// SoftMesh_74036646248987
// MI455X (gfx1250) — compile-verified, matrix operand order checked by reading, not run on hardware
//
#include <hip/hip_runtime.h>

// MLP forward: [N,3] -> 50 -> (7x 50->50 tanh) -> 3.
// D = A(W^T) x B(h): activations live in registers as WMMA B-fragments for the
// whole network; D-tile layout maps element-for-element onto the next layer's
// B-fragment layout. Biases ride a constant-1 activation column at k=50.
// Layer loop is kept ROLLED so weight-fragment LDS loads cannot be hoisted
// out of the grid-stride tile loop (round-2 showed that hoisting -> 448 VGPRs
// of scratch spills). Each wave processes TWO 16-point tiles so every weight
// fragment load feeds two WMMAs.

#define HDIM  50
#define HP    64
#define NMID  7
#define WAVES 8
#define BLOCK (WAVES * 32)

typedef _Float16 v16h __attribute__((ext_vector_type(16)));
typedef _Float16 v8h  __attribute__((ext_vector_type(8)));
typedef float    v8f  __attribute__((ext_vector_type(8)));
typedef float    v2f  __attribute__((ext_vector_type(2)));

union Frag16 { v16h v; v8h h[2]; };

#if __has_builtin(__builtin_amdgcn_wmma_f32_16x16x4_f32)
#define HAVE_WMMA4 1
#else
#define HAVE_WMMA4 0
#endif

__device__ __forceinline__ float fast_tanh(float x) {
#if __has_builtin(__builtin_amdgcn_tanhf)
    return __builtin_amdgcn_tanhf(x);            // v_tanh_f32 on gfx1250
#else
    float e = __builtin_amdgcn_exp2f(x * 2.8853900817779268f);
    return 1.0f - 2.0f / (e + 1.0f);
#endif
}

__device__ __forceinline__ v8f wmma16(const Frag16& a, const Frag16& b, v8f c) {
    return __builtin_amdgcn_wmma_f32_16x16x32_f16(
        false, a.v, false, b.v, (short)0, c, false, false);
}

__global__ __launch_bounds__(BLOCK)
void mlp_pinn_fwd(const float* __restrict__ x, const float* __restrict__ y,
                  const float* __restrict__ z,
                  const float* __restrict__ Win,  const float* __restrict__ bin,
                  const float* __restrict__ Wmid, const float* __restrict__ bmid,
                  const float* __restrict__ Wout, const float* __restrict__ bout,
                  float* __restrict__ out, int n)
{
    // wt[layer][m][k] = Wmid[layer][k][m]; column k==50 holds bias.
    __shared__ __align__(16) _Float16 wt[NMID * HP * HP];   // 56 KB
    __shared__ __align__(16) _Float16 wto[16 * HP];         // [Wout | bout]^T
    __shared__ __align__(16) float    swin4[HP * 4];        // [Win | bin]^T, K=4

    const int tid = threadIdx.x;

    for (int idx = tid; idx < NMID * HP * HP; idx += BLOCK) {
        int i = idx >> 12, rem = idx & 4095;
        int k = rem >> 6,  m = rem & 63;
        float v = 0.0f;
        if (m < HDIM) {
            if (k < HDIM)       v = Wmid[(i * HDIM + k) * HDIM + m];
            else if (k == HDIM) v = bmid[i * HDIM + m];
        }
        wt[i * HP * HP + m * HP + k] = (_Float16)v;
    }
    for (int idx = tid; idx < 16 * HP; idx += BLOCK) {
        int m = idx >> 6, k = idx & 63;
        float v = 0.0f;
        if (m < 3) {
            if (k < HDIM)       v = Wout[k * 3 + m];
            else if (k == HDIM) v = bout[m];
        }
        wto[idx] = (_Float16)v;
    }
    for (int idx = tid; idx < HP * 4; idx += BLOCK) {
        int m = idx >> 2, k = idx & 3;
        float v = 0.0f;
        if (m < HDIM) v = (k < 3) ? Win[k * HDIM + m] : bin[m];
        swin4[idx] = v;
    }
    __syncthreads();

    const int lane  = tid & 31;
    const int wave  = tid >> 5;
    const int laneL = lane & 15;    // point within tile (B/D column)
    const int half  = lane >> 4;    // K sub-block selector

    const int nst        = (n + 31) >> 5;           // 32-point super-tiles
    const int gwave      = blockIdx.x * WAVES + wave;
    const int waveStride = gridDim.x * WAVES;

    for (int st = gwave; st < nst; st += waveStride) {
        const int gpA = st * 32 + laneL;
        const int gpB = gpA + 16;

        float xA = 0.f, yA = 0.f, zA = 0.f, xB = 0.f, yB = 0.f, zB = 0.f;
        if (gpA < n) { xA = x[gpA]; yA = y[gpA]; zA = z[gpA]; }
        if (gpB < n) { xB = x[gpB]; yB = y[gpB]; zB = z[gpB]; }

        // Current-layer activations as B fragments (2 tiles x 2 K-chunks).
        Frag16 bA0, bA1, bB0, bB1;

        // ---- input layer ----
#if HAVE_WMMA4
        {
            // B (4x16 f32): VGPR0 = K(2*half), VGPR1 = K(2*half+1); K order x,y,z,1.
            v2f biA, biB;
            biA[0] = half ? zA : xA;  biA[1] = half ? 1.0f : yA;
            biB[0] = half ? zB : xB;  biB[1] = half ? 1.0f : yB;
            #pragma unroll
            for (int mt = 0; mt < 4; ++mt) {
                const float* wr = &swin4[(mt * 16 + laneL) * 4 + 2 * half];
                v2f a; a[0] = wr[0]; a[1] = wr[1];
                v8f cA = {}, cB = {};
                cA = __builtin_amdgcn_wmma_f32_16x16x4_f32(
                         false, a, false, biA, (short)0, cA, false, false);
                cB = __builtin_amdgcn_wmma_f32_16x16x4_f32(
                         false, a, false, biB, (short)0, cB, false, false);
                #pragma unroll
                for (int r = 0; r < 8; ++r) {
                    int e = r + 8 * (mt & 1);
                    _Float16 tA = (_Float16)fast_tanh(cA[r]);
                    _Float16 tB = (_Float16)fast_tanh(cB[r]);
                    if (mt < 2) { bA0.v[e] = tA; bB0.v[e] = tB; }
                    else        { bA1.v[e] = tA; bB1.v[e] = tB; }
                }
            }
        }
#else
        #pragma unroll
        for (int kc = 0; kc < 2; ++kc) {
            #pragma unroll
            for (int e = 0; e < 16; ++e) {
                int f = 32 * kc + 16 * (e >> 3) + (e & 7) + 8 * half;
                const float* wr = &swin4[f * 4];
                float vA = fmaf(xA, wr[0], fmaf(yA, wr[1], fmaf(zA, wr[2], wr[3])));
                float vB = fmaf(xB, wr[0], fmaf(yB, wr[1], fmaf(zB, wr[2], wr[3])));
                _Float16 tA = (_Float16)fast_tanh(vA);
                _Float16 tB = (_Float16)fast_tanh(vB);
                if (kc == 0) { bA0.v[e] = tA; bB0.v[e] = tB; }
                else         { bA1.v[e] = tA; bB1.v[e] = tB; }
            }
        }
#endif
        // constant-1 bias column (k=50: half==0 lanes, element 10 of chunk 1)
        bA1.v[10] = half ? bA1.v[10] : (_Float16)1.0f;
        bB1.v[10] = half ? bB1.v[10] : (_Float16)1.0f;

        // ---- 7 hidden layers; ROLLED so weight loads can't be hoisted ----
        #pragma clang loop unroll(disable)
        for (int layer = 0; layer < NMID; ++layer) {
            const _Float16* wl = &wt[layer * HP * HP];
            Frag16 nA0, nA1, nB0, nB1;
            #pragma unroll
            for (int mt = 0; mt < 4; ++mt) {
                const _Float16* wrow = wl + (mt * 16 + laneL) * HP;
                Frag16 a0, a1;
                a0.h[0] = *(const v8h*)(wrow +      8 * half);
                a0.h[1] = *(const v8h*)(wrow + 16 + 8 * half);
                a1.h[0] = *(const v8h*)(wrow + 32 + 8 * half);
                a1.h[1] = *(const v8h*)(wrow + 48 + 8 * half);
                v8f cA = {}, cB = {};
                cA = wmma16(a0, bA0, cA); cA = wmma16(a1, bA1, cA);
                cB = wmma16(a0, bB0, cB); cB = wmma16(a1, bB1, cB);
                #pragma unroll
                for (int r = 0; r < 8; ++r) {
                    int e = r + 8 * (mt & 1);
                    _Float16 tA = (_Float16)fast_tanh(cA[r]);
                    _Float16 tB = (_Float16)fast_tanh(cB[r]);
                    if (mt < 2) { nA0.v[e] = tA; nB0.v[e] = tB; }
                    else        { nA1.v[e] = tA; nB1.v[e] = tB; }
                }
            }
            nA1.v[10] = half ? nA1.v[10] : (_Float16)1.0f;
            nB1.v[10] = half ? nB1.v[10] : (_Float16)1.0f;
            bA0 = nA0; bA1 = nA1; bB0 = nB0; bB1 = nB1;
        }

        // ---- output layer: [Wout|bout]^T rows 0..2; U[o] at half==0, elem o ----
        {
            const _Float16* worow = &wto[laneL * HP];
            Frag16 a0, a1;
            a0.h[0] = *(const v8h*)(worow +      8 * half);
            a0.h[1] = *(const v8h*)(worow + 16 + 8 * half);
            a1.h[0] = *(const v8h*)(worow + 32 + 8 * half);
            a1.h[1] = *(const v8h*)(worow + 48 + 8 * half);
            v8f cA = {}, cB = {};
            cA = wmma16(a0, bA0, cA); cA = wmma16(a1, bA1, cA);
            cB = wmma16(a0, bB0, cB); cB = wmma16(a1, bB1, cB);
            if (half == 0) {
                if (gpA < n) {
                    out[0 * (size_t)n + gpA] = cA[0];
                    out[1 * (size_t)n + gpA] = cA[1];
                    out[2 * (size_t)n + gpA] = cA[2];
                }
                if (gpB < n) {
                    out[0 * (size_t)n + gpB] = cB[0];
                    out[1 * (size_t)n + gpB] = cB[1];
                    out[2 * (size_t)n + gpB] = cB[2];
                }
            }
        }
    }
}

extern "C" void kernel_launch(void* const* d_in, const int* in_sizes, int n_in,
                              void* d_out, int out_size, void* d_ws, size_t ws_size,
                              hipStream_t stream) {
    const float* x    = (const float*)d_in[0];
    const float* y    = (const float*)d_in[1];
    const float* z    = (const float*)d_in[2];
    const float* Win  = (const float*)d_in[3];
    const float* bin  = (const float*)d_in[4];
    const float* Wmid = (const float*)d_in[5];
    const float* bmid = (const float*)d_in[6];
    const float* Wout = (const float*)d_in[7];
    const float* bout = (const float*)d_in[8];
    float* out = (float*)d_out;

    int n      = in_sizes[0];
    int nst    = (n + 31) / 32;
    int blocks = (nst + WAVES - 1) / WAVES;
    if (blocks > 1536) blocks = 1536;
    if (blocks < 1) blocks = 1;
    mlp_pinn_fwd<<<blocks, BLOCK, 0, stream>>>(x, y, z, Win, bin, Wmid, bmid,
                                               Wout, bout, out, n);
}
